// ExpertPool_78288663872347
// MI455X (gfx1250) — matrix operand order repaired and checked by reading, not compiled, not run on hardware
//
#include <hip/hip_runtime.h>
#include <math.h>

#define T_TOK 2048
#define DD 768
#define HH 3072
#define EE 8
#define MAXENT 4096   // T*K upper bound on dispatched (token,expert) pairs

typedef __attribute__((ext_vector_type(16))) __bf16 v16bf;
typedef __attribute__((ext_vector_type(8)))  __bf16 v8bf;
typedef __attribute__((ext_vector_type(8)))  float  v8f;

__device__ __forceinline__ unsigned short f2bf(float f){
  union { float f; unsigned int u; } x; x.f = f;
  unsigned int r = x.u + 0x7FFFu + ((x.u >> 16) & 1u);   // round-to-nearest-even
  return (unsigned short)(r >> 16);
}

__device__ __forceinline__ v8f wmma_bf16(v16bf a, v16bf b, v8f c){
  return __builtin_amdgcn_wmma_f32_16x16x32_bf16(false, a, false, b, (short)0, c, false, false);
}

__device__ __forceinline__ v16bf combine16(v8bf lo, v8bf hi){
  return __builtin_shufflevector(lo, hi, 0,1,2,3,4,5,6,7,8,9,10,11,12,13,14,15);
}

// A fragment (16x32 bf16, M x K) from a row-major array, leading dim ld (elements).
// Lane L<16: row base+L, K = k0+{0..7, 16..23}; lane L+16: same row, K = k0+{8..15, 24..31}.
__device__ __forceinline__ v16bf load_a(const unsigned short* base, int ld, int row, int k0, int lane){
  int r  = row + (lane & 15);
  int ko = k0 + ((lane >> 4) << 3);
  const unsigned short* p = base + (size_t)r * ld + ko;
  v8bf lo = *(const v8bf*)(p);
  v8bf hi = *(const v8bf*)(p + 16);
  return combine16(lo, hi);
}

__device__ __forceinline__ v16bf load_a_clamp(const unsigned short* base, int ld, int row, int maxrow, int k0, int lane){
  int r  = row + (lane & 15); if (r > maxrow) r = maxrow;
  int ko = k0 + ((lane >> 4) << 3);
  const unsigned short* p = base + (size_t)r * ld + ko;
  v8bf lo = *(const v8bf*)(p);
  v8bf hi = *(const v8bf*)(p + 16);
  return combine16(lo, hi);
}

// B fragment (32x16 bf16, K x N) from row-major weights W[out][in] (ldk = in-dim).
// Lane L<16: column n+L, K = k0+{0..7, 16..23}; lane L+16: K = k0+{8..15, 24..31}.
__device__ __forceinline__ v16bf load_b(const unsigned short* W, int ldk, int n, int k0, int lane){
  int col = n + (lane & 15);
  int ko  = k0 + ((lane >> 4) << 3);
  const unsigned short* p = W + (size_t)col * ldk + ko;
  v8bf lo = *(const v8bf*)(p);
  v8bf hi = *(const v8bf*)(p + 16);
  return combine16(lo, hi);
}

// Prefetch (global_prefetch_b8) one cacheline per output column, `k` elements into the row.
__device__ __forceinline__ void prefetch_rows(const unsigned short* W, int ldk, int col0, int k, int lane){
  __builtin_prefetch((const void*)(W + (size_t)(col0 + lane) * ldk + k), 0, 1);
}

// ---------------- f32 -> bf16 conversion ----------------
__global__ void cvt_kernel(const float* __restrict__ src, unsigned short* __restrict__ dst, int n4){
  int i = blockIdx.x * blockDim.x + threadIdx.x;
  int stride = gridDim.x * blockDim.x;
  for (; i < n4; i += stride){
    float4 v = ((const float4*)src)[i];
    ushort4 o;
    o.x = f2bf(v.x); o.y = f2bf(v.y); o.z = f2bf(v.z); o.w = f2bf(v.w);
    ((ushort4*)dst)[i] = o;
  }
}

// ---------------- deterministic routing ----------------
__global__ void route_kernel(const float* __restrict__ comb,
                             int* __restrict__ counts, int* __restrict__ offsets,
                             int* __restrict__ list, int* __restrict__ pos){
  int e = threadIdx.x;
  if (e < EE){
    int c = 0;
    for (int t = 0; t < T_TOK; ++t){
      float w = comb[t * EE + e];
      if (w > 0.f){ list[e * T_TOK + c] = t; pos[t * EE + e] = c; ++c; }
      else         { pos[t * EE + e] = -1; }
    }
    counts[e] = c;
  }
  __syncthreads();
  if (threadIdx.x == 0){
    int off = 0;
    for (int k = 0; k < EE; ++k){ offsets[k] = off; off += counts[k]; }
    offsets[EE] = off;
  }
}

// ---------------- stage 1: h = gelu(x Wg^T) * (x Wv^T) ----------------
#define CHK 192
#define LDA 200   // pad: row stride 100 dwords -> uniform 2-way LDS banking for b128 reads
__global__ __launch_bounds__(256) void ffn1_kernel(
    const unsigned short* __restrict__ tok,
    const unsigned short* __restrict__ Wg,
    const unsigned short* __restrict__ Wv,
    const int* __restrict__ counts, const int* __restrict__ offsets,
    const int* __restrict__ list,
    unsigned short* __restrict__ Xh)
{
  const int e = blockIdx.z;
  const int count = counts[e];
  const int m0 = blockIdx.y * 64;
  if (m0 >= count) return;
  const int n0 = blockIdx.x * 128;

  __shared__ unsigned short As[64 * LDA];
  __shared__ int rowtok[64];

  const int tid = threadIdx.x;
  if (tid < 64){
    int r = m0 + tid;
    rowtok[tid] = (r < count) ? list[e * T_TOK + r] : -1;
  }
  const int wave = tid >> 5, lane = tid & 31;
  const int wm = wave & 1, wn = wave >> 1;          // wave tile: rows 32*wm, cols 32*wn
  const unsigned short* Wge = Wg + (size_t)e * HH * DD;
  const unsigned short* Wve = Wv + (size_t)e * HH * DD;

  v8f cg[2][2] = {}; v8f cv[2][2] = {};

  for (int kc = 0; kc < DD; kc += CHK){
    __syncthreads();
    // gather 64 token rows x CHK cols into LDS (16B chunks)
    for (int c = tid; c < 64 * (CHK / 8); c += 256){
      int r  = c / (CHK / 8);
      int cc = (c % (CHK / 8)) * 8;
      int t  = rowtok[r];
      uint4 v = make_uint4(0u, 0u, 0u, 0u);
      if (t >= 0) v = *(const uint4*)(tok + (size_t)t * DD + kc + cc);
      *(uint4*)(As + r * LDA + cc) = v;
    }
    __syncthreads();
    for (int kk = 0; kk < CHK; kk += 32){
      v16bf a0 = load_a(As, LDA, wm * 32,      kk, lane);
      v16bf a1 = load_a(As, LDA, wm * 32 + 16, kk, lane);
      int k0 = kc + kk;
      int kpf = k0 + 256;                 // 8 k-steps ahead
      if (kpf < DD){
        prefetch_rows(Wge, DD, n0 + wn * 32, kpf, lane);
        prefetch_rows(Wve, DD, n0 + wn * 32, kpf, lane);
      }
      v16bf bg0 = load_b(Wge, DD, n0 + wn * 32,      k0, lane);
      v16bf bg1 = load_b(Wge, DD, n0 + wn * 32 + 16, k0, lane);
      v16bf bv0 = load_b(Wve, DD, n0 + wn * 32,      k0, lane);
      v16bf bv1 = load_b(Wve, DD, n0 + wn * 32 + 16, k0, lane);
      cg[0][0] = wmma_bf16(a0, bg0, cg[0][0]);
      cg[0][1] = wmma_bf16(a0, bg1, cg[0][1]);
      cg[1][0] = wmma_bf16(a1, bg0, cg[1][0]);
      cg[1][1] = wmma_bf16(a1, bg1, cg[1][1]);
      cv[0][0] = wmma_bf16(a0, bv0, cv[0][0]);
      cv[0][1] = wmma_bf16(a0, bv1, cv[0][1]);
      cv[1][0] = wmma_bf16(a1, bv0, cv[1][0]);
      cv[1][1] = wmma_bf16(a1, bv1, cv[1][1]);
    }
  }

  const int gbase = offsets[e];
  const int hi8 = (lane >> 4) << 3;
  const int nn  = lane & 15;
  for (int mi = 0; mi < 2; ++mi)
    for (int ni = 0; ni < 2; ++ni)
      #pragma unroll
      for (int i = 0; i < 8; ++i){
        int row = m0 + wm * 32 + mi * 16 + hi8 + i;
        if (row < count){
          float g = cg[mi][ni][i];
          float h = 0.5f * g * (1.0f + erff(g * 0.70710678118654752f)) * cv[mi][ni][i];
          int col = n0 + wn * 32 + ni * 16 + nn;
          Xh[(size_t)(gbase + row) * HH + col] = f2bf(h);
        }
      }
}

// ---------------- stage 2: expert_out = h Wo^T * scale ----------------
__global__ __launch_bounds__(256) void ffn2_kernel(
    const unsigned short* __restrict__ Xh,
    const unsigned short* __restrict__ Wo,
    const int* __restrict__ counts, const int* __restrict__ offsets,
    const float* __restrict__ scale,
    float* __restrict__ Eout)
{
  const int e = blockIdx.z;
  const int count = counts[e];
  const int m0 = blockIdx.y * 64;
  if (m0 >= count) return;
  const int n0 = blockIdx.x * 256;
  const int tid = threadIdx.x, wave = tid >> 5, lane = tid & 31;
  const int wm = wave & 1, wn = wave >> 1;          // wave tile: 32 rows x 64 cols
  const int gbase = offsets[e];
  const unsigned short* Ae = Xh + (size_t)gbase * HH;
  const unsigned short* We = Wo + (size_t)e * DD * HH;
  const int maxrow = count - 1;

  v8f c[2][4] = {};
  for (int k0 = 0; k0 < HH; k0 += 32){
    int kpf = k0 + 512;                   // 16 k-steps ahead on the long K loop
    if (kpf < HH){
      prefetch_rows(We, HH, n0 + wn * 64,      kpf, lane);
      prefetch_rows(We, HH, n0 + wn * 64 + 32, kpf, lane);
    }
    v16bf a0 = load_a_clamp(Ae, HH, m0 + wm * 32,      maxrow, k0, lane);
    v16bf a1 = load_a_clamp(Ae, HH, m0 + wm * 32 + 16, maxrow, k0, lane);
    v16bf b0 = load_b(We, HH, n0 + wn * 64,      k0, lane);
    v16bf b1 = load_b(We, HH, n0 + wn * 64 + 16, k0, lane);
    v16bf b2 = load_b(We, HH, n0 + wn * 64 + 32, k0, lane);
    v16bf b3 = load_b(We, HH, n0 + wn * 64 + 48, k0, lane);
    c[0][0] = wmma_bf16(a0, b0, c[0][0]);
    c[0][1] = wmma_bf16(a0, b1, c[0][1]);
    c[0][2] = wmma_bf16(a0, b2, c[0][2]);
    c[0][3] = wmma_bf16(a0, b3, c[0][3]);
    c[1][0] = wmma_bf16(a1, b0, c[1][0]);
    c[1][1] = wmma_bf16(a1, b1, c[1][1]);
    c[1][2] = wmma_bf16(a1, b2, c[1][2]);
    c[1][3] = wmma_bf16(a1, b3, c[1][3]);
  }

  const float sc = scale[e];
  const int hi8 = (lane >> 4) << 3;
  const int nn  = lane & 15;
  for (int mi = 0; mi < 2; ++mi)
    for (int ni = 0; ni < 4; ++ni)
      #pragma unroll
      for (int i = 0; i < 8; ++i){
        int row = m0 + wm * 32 + mi * 16 + hi8 + i;
        if (row < count){
          int col = n0 + wn * 64 + ni * 16 + nn;
          Eout[(size_t)(gbase + row) * DD + col] = c[mi][ni][i] * sc;
        }
      }
}

// ---------------- deterministic weighted combine ----------------
__global__ void combine_kernel(const float* __restrict__ comb,
                               const float* __restrict__ Eout,
                               const int* __restrict__ offsets,
                               const int* __restrict__ pos,
                               float* __restrict__ y)
{
  int i = blockIdx.x * blockDim.x + threadIdx.x;
  int stride = gridDim.x * blockDim.x;
  for (; i < T_TOK * DD; i += stride){
    int t = i / DD, d = i - t * DD;
    float acc = 0.f;
    #pragma unroll
    for (int e = 0; e < EE; ++e){
      int p = pos[t * EE + e];
      if (p >= 0) acc += comb[t * EE + e] * Eout[(size_t)(offsets[e] + p) * DD + d];
    }
    y[i] = acc;
  }
}

extern "C" void kernel_launch(void* const* d_in, const int* in_sizes, int n_in,
                              void* d_out, int out_size, void* d_ws, size_t ws_size,
                              hipStream_t stream)
{
  (void)in_sizes; (void)n_in; (void)out_size; (void)ws_size;
  const float* tokens = (const float*)d_in[0];
  const float* comb   = (const float*)d_in[2];
  const float* Wg     = (const float*)d_in[3];
  const float* Wv     = (const float*)d_in[4];
  const float* Wo     = (const float*)d_in[5];
  const float* scale  = (const float*)d_in[6];
  float* y = (float*)d_out;

  char* ws = (char*)d_ws;
  size_t off = 0;
  auto alloc = [&](size_t bytes)->char*{
    char* p = ws + off;
    off = (off + bytes + 255) & ~(size_t)255;
    return p;
  };
  unsigned short* tok_bf = (unsigned short*)alloc((size_t)T_TOK * DD * 2);
  unsigned short* Wg_bf  = (unsigned short*)alloc((size_t)EE * HH * DD * 2);
  unsigned short* Wv_bf  = (unsigned short*)alloc((size_t)EE * HH * DD * 2);
  unsigned short* Wo_bf  = (unsigned short*)alloc((size_t)EE * HH * DD * 2);
  unsigned short* Xh     = (unsigned short*)alloc((size_t)MAXENT * HH * 2);
  float*          Eout   = (float*)alloc((size_t)MAXENT * DD * 4);
  int* counts  = (int*)alloc(EE * 4);
  int* offsets = (int*)alloc((EE + 1) * 4);
  int* list    = (int*)alloc((size_t)EE * T_TOK * 4);
  int* pos     = (int*)alloc((size_t)T_TOK * EE * 4);

  cvt_kernel<<<1024, 256, 0, stream>>>(tokens, tok_bf, T_TOK * DD / 4);
  cvt_kernel<<<2048, 256, 0, stream>>>(Wg, Wg_bf, EE * HH * DD / 4);
  cvt_kernel<<<2048, 256, 0, stream>>>(Wv, Wv_bf, EE * HH * DD / 4);
  cvt_kernel<<<2048, 256, 0, stream>>>(Wo, Wo_bf, EE * HH * DD / 4);
  route_kernel<<<1, 64, 0, stream>>>(comb, counts, offsets, list, pos);
  ffn1_kernel<<<dim3(HH / 128, (T_TOK + 63) / 64, EE), 256, 0, stream>>>(
      tok_bf, Wg_bf, Wv_bf, counts, offsets, list, Xh);
  ffn2_kernel<<<dim3(DD / 256, (T_TOK + 63) / 64, EE), 256, 0, stream>>>(
      Xh, Wo_bf, counts, offsets, scale, Eout);
  combine_kernel<<<2048, 256, 0, stream>>>(comb, Eout, offsets, pos, y);
}
